// Block_80839874445488
// MI455X (gfx1250) — compile-verified, matrix operand order checked by reading, not run on hardware
//
#include <hip/hip_runtime.h>
#include <math.h>

typedef __attribute__((ext_vector_type(16))) _Float16 v16h;
typedef __attribute__((ext_vector_type(8)))  float    v8f;

// ---------------- problem constants ----------------
constexpr int BB    = 8;
constexpr int NN    = 1025;
constexpr int CC    = 1024;
constexpr int HH    = 16;
constexpr int DHD   = 64;
constexpr int HIDD  = 4096;
constexpr int MROWS = BB * NN;          // 8200
constexpr int BH    = BB * HH;          // 128
constexpr int NPADV = 1056;             // 33 * 32, padded K for AV
constexpr int SR    = BH * NN;          // softmax rows = 131200
constexpr float EPSF  = 1e-5f;
constexpr float NEGF  = -65504.0f;
constexpr float SCALE = 0.125f;         // DH^-0.5
constexpr int LDS_STRIDE = 40;          // halves; 80B rows: 16B aligned, conflict-free
constexpr int TILE_HALVES = 128 * LDS_STRIDE;   // 5120 halves = 10240 bytes / tile

#ifndef __has_builtin
#define __has_builtin(x) 0
#endif
#if __has_builtin(__builtin_amdgcn_tensor_load_to_lds) && \
    __has_builtin(__builtin_amdgcn_s_wait_tensorcnt)
#define USE_TDM 1
#else
#define USE_TDM 0
#endif

// ---------------- WMMA helpers ----------------
static __device__ __forceinline__ v8f wmma16(v16h a, v16h b, v8f c) {
  // D = A(16x32 f16) * B(32x16 f16) + C(16x16 f32)
  return __builtin_amdgcn_wmma_f32_16x16x32_f16(false, a, false, b, (short)0, c,
                                                false, false);
}

// Fragment load from a row-major [16][32] fp16 tile in LDS (stride 40 halves).
// 16-bit A layout: lanes 0-15 -> K 0..7 & 16..23, lanes 16-31 -> K 8..15 & 24..31.
static __device__ __forceinline__ v16h lds_frag(const _Float16* tile, int lane) {
  const int r = lane & 15;
  const int k = (lane >> 4) << 3;
  const _Float16* p = tile + r * LDS_STRIDE + k;
  union { uint4 u[2]; v16h v; } u;
  u.u[0] = *(const uint4*)(p);
  u.u[1] = *(const uint4*)(p + 16);
  return u.v;
}

// Same fragment pattern, direct from global row-major [rows][64] fp16, row clamped.
static __device__ __forceinline__ v16h gfrag64(const _Float16* base, int row0,
                                               int kOff, int lane) {
  int r = row0 + (lane & 15);
  if (r > NN - 1) r = NN - 1;
  const _Float16* p = base + (size_t)r * 64 + kOff + ((lane >> 4) << 3);
  union { uint4 u[2]; v16h v; } u;
  u.u[0] = *(const uint4*)(p);
  u.u[1] = *(const uint4*)(p + 16);
  return u.v;
}

#if USE_TDM
// ---------------- Tensor Data Mover: 2D tile (tile_dim1=128 rows x 32 halves) ----
// D# packing per CDNA5 ISA:
//  group0: [1:0]count=1, [63:32]lds_addr, [120:64]global_addr, [127:126]type=2
//  group1: data_size=1(2B), pad_enable, pad_interval=3 (16 DW = one 32-half row),
//          pad_amount=3 (4 DW = 8 halves)  -> LDS row stride 40 halves,
//          tensor_dim0=k_rem, tensor_dim1=rows_rem (OOB rows read as zero),
//          tile_dim0=32, tile_dim1=128, tensor_dim0_stride=K.
typedef unsigned int u32x4 __attribute__((ext_vector_type(4)));
typedef int i32x4 __attribute__((ext_vector_type(4)));
typedef int i32x8 __attribute__((ext_vector_type(8)));

static __device__ __forceinline__ void tdm_load_2d(unsigned lds_off,
                                                   const _Float16* gptr,
                                                   int rows, int kElems,
                                                   int strideElems) {
  const unsigned long long ga = (unsigned long long)(size_t)gptr;
  const u32x4 g0 = {1u, lds_off, (unsigned)ga,
                    (unsigned)((ga >> 32) & 0x1FFFFFFull) | (2u << 30)};
  const unsigned flags = (1u << 16) | (1u << 20) | (3u << 22) | (3u << 25);
  const i32x8 g1 = {
      (int)flags,
      (int)(((unsigned)kElems & 0xFFFFu) << 16),
      (int)(((unsigned)kElems >> 16) | (((unsigned)rows & 0xFFFFu) << 16)),
      (int)(((unsigned)rows >> 16) | (32u << 16)),
      (int)128,
      (int)strideElems,
      0, 0};
  const i32x4 gz = {0, 0, 0, 0};
#if defined(__clang_major__) && __clang_major__ >= 23
  const i32x8 gz8 = {0, 0, 0, 0, 0, 0, 0, 0};
  __builtin_amdgcn_tensor_load_to_lds(g0, g1, gz, gz, gz8, 0);
#else
  __builtin_amdgcn_tensor_load_to_lds(g0, g1, gz, gz, 0);
#endif
}
#endif  // USE_TDM

// ---------------- weight transpose + fp16 convert: Wt[n][k] = W[k][n] ----------------
__global__ __launch_bounds__(256) void wtrans(const float* __restrict__ W,
                                              _Float16* __restrict__ Wt,
                                              int K, int Nn) {
  size_t idx = (size_t)blockIdx.x * 256 + threadIdx.x;
  if (idx >= (size_t)K * Nn) return;
  int n = (int)(idx % Nn);
  int k = (int)(idx / Nn);
  Wt[(size_t)n * K + k] = (_Float16)W[idx];
}

// ---------------- LayerNorm (one row per block) -> fp16 ----------------
__global__ __launch_bounds__(256) void layernorm_k(const float* __restrict__ x,
                                                   const float* __restrict__ g,
                                                   const float* __restrict__ bt,
                                                   _Float16* __restrict__ out) {
  __shared__ float s1[256], s2[256];
  const int row = blockIdx.x, t = threadIdx.x;
  const float* xr = x + (size_t)row * CC;
  float4 v = *(const float4*)(xr + t * 4);
  s1[t] = v.x + v.y + v.z + v.w;
  s2[t] = v.x * v.x + v.y * v.y + v.z * v.z + v.w * v.w;
  __syncthreads();
  for (int st = 128; st > 0; st >>= 1) {
    if (t < st) { s1[t] += s1[t + st]; s2[t] += s2[t + st]; }
    __syncthreads();
  }
  const float mu = s1[0] * (1.0f / CC);
  const float var = s2[0] * (1.0f / CC) - mu * mu;
  const float rs = rsqrtf(var + EPSF);
  const int c0 = t * 4;
  union { _Float16 h[4]; uint2 u; } o;
  o.h[0] = (_Float16)((v.x - mu) * rs * g[c0 + 0] + bt[c0 + 0]);
  o.h[1] = (_Float16)((v.y - mu) * rs * g[c0 + 1] + bt[c0 + 1]);
  o.h[2] = (_Float16)((v.z - mu) * rs * g[c0 + 2] + bt[c0 + 2]);
  o.h[3] = (_Float16)((v.w - mu) * rs * g[c0 + 3] + bt[c0 + 3]);
  *(uint2*)(out + (size_t)row * CC + c0) = o.u;
}

// ---------------- generic WMMA GEMM: C = A[M,K] * Bt[N,K]^T + bias ----------------
// EPI 0: fp16 out.  EPI 1: f32 out = c + bias + resid.  EPI 2: fp16 gelu(c+bias).
// Dynamic LDS: 4 x 10240B tile buffers (A0,B0,A1,B1) -> TDM double buffering.
template <int EPI>
__global__ __launch_bounds__(256) void gemm16(const _Float16* __restrict__ A,
                                              const _Float16* __restrict__ Bt,
                                              const float* __restrict__ bias,
                                              const float* __restrict__ resid,
                                              _Float16* __restrict__ outH,
                                              float* __restrict__ outF,
                                              int Mr, int Nn, int K) {
  extern __shared__ _Float16 smem[];
  const int tid = threadIdx.x, lane = tid & 31, wave = tid >> 5;
  const int wm = wave >> 2, wn = wave & 3;      // 2 x 4 wave grid
  const int m0 = blockIdx.y * 128, n0 = blockIdx.x * 128;
  v8f acc[4][2] = {};

#if USE_TDM
  const bool issuer = (wave == 0);
  const unsigned tileBytes = (unsigned)(TILE_HALVES * 2);
  if (issuer) {                                  // preload slice 0 via TDM
    tdm_load_2d(0u, A + (size_t)m0 * K, Mr - m0, K, K);
    tdm_load_2d(tileBytes, Bt + (size_t)n0 * K, Nn - n0, K, K);
  }
  int p = 0;
  for (int k0 = 0; k0 < K; k0 += 32) {
    __syncthreads();                             // buf[p^1] free for next DMA
    const bool more = (k0 + 32) < K;
    if (issuer) {
      if (more) {                                // overlap next DMA with compute
        const unsigned nb = (unsigned)((p ^ 1) * 2) * tileBytes;
        tdm_load_2d(nb, A + (size_t)m0 * K + k0 + 32, Mr - m0, K - k0 - 32, K);
        tdm_load_2d(nb + tileBytes, Bt + (size_t)n0 * K + k0 + 32, Nn - n0,
                    K - k0 - 32, K);
        __builtin_amdgcn_s_wait_tensorcnt(2);    // current slice complete
      } else {
        __builtin_amdgcn_s_wait_tensorcnt(0);
      }
    }
    __syncthreads();                             // broadcast tile readiness
    _Float16* ap = smem + (unsigned)(p * 2) * (unsigned)TILE_HALVES;
    _Float16* bp = ap + TILE_HALVES;
    const v16h bf0 = lds_frag(bp + (wn * 32 + 0) * LDS_STRIDE, lane);
    const v16h bf1 = lds_frag(bp + (wn * 32 + 16) * LDS_STRIDE, lane);
#pragma unroll
    for (int i = 0; i < 4; ++i) {
      const v16h af = lds_frag(ap + (wm * 64 + i * 16) * LDS_STRIDE, lane);
      acc[i][0] = wmma16(af, bf0, acc[i][0]);
      acc[i][1] = wmma16(af, bf1, acc[i][1]);
    }
    p ^= 1;
  }
#else
  _Float16* sA = smem;
  _Float16* sB = smem + TILE_HALVES;
  for (int k0 = 0; k0 < K; k0 += 32) {
#pragma unroll
    for (int i = 0; i < 2; ++i) {                // stage 128x32 A and Bt tiles
      const int c = tid + 256 * i;
      const int r = c >> 2, s = c & 3;
      int gr = m0 + r; if (gr > Mr - 1) gr = Mr - 1;
      int gc = n0 + r; if (gc > Nn - 1) gc = Nn - 1;
      const size_t ao = (size_t)gr * K + k0 + s * 8;
      const size_t bo = (size_t)gc * K + k0 + s * 8;
      *(uint4*)(sA + r * LDS_STRIDE + s * 8) = *(const uint4*)(A + ao);
      *(uint4*)(sB + r * LDS_STRIDE + s * 8) = *(const uint4*)(Bt + bo);
      if (k0 + 32 < K) {
        __builtin_prefetch(A + ao + 32, 0, 3);
        __builtin_prefetch(Bt + bo + 32, 0, 3);
      }
    }
    __syncthreads();
    const v16h bf0 = lds_frag(sB + (wn * 32 + 0) * LDS_STRIDE, lane);
    const v16h bf1 = lds_frag(sB + (wn * 32 + 16) * LDS_STRIDE, lane);
#pragma unroll
    for (int i = 0; i < 4; ++i) {
      const v16h af = lds_frag(sA + (wm * 64 + i * 16) * LDS_STRIDE, lane);
      acc[i][0] = wmma16(af, bf0, acc[i][0]);
      acc[i][1] = wmma16(af, bf1, acc[i][1]);
    }
    __syncthreads();
  }
#endif

#pragma unroll
  for (int i = 0; i < 4; ++i)
#pragma unroll
    for (int j = 0; j < 2; ++j) {
      const int col = n0 + wn * 32 + j * 16 + (lane & 15);
      const float bv = bias[col];
      const int rbase = m0 + wm * 64 + i * 16 + ((lane >> 4) << 3);
#pragma unroll
      for (int r = 0; r < 8; ++r) {
        const int row = rbase + r;
        if (row < Mr) {
          const float v = acc[i][j][r] + bv;
          const size_t o = (size_t)row * Nn + col;
          if (EPI == 0) {
            outH[o] = (_Float16)v;
          } else if (EPI == 1) {
            outF[o] = v + resid[o];
          } else {
            outH[o] = (_Float16)(0.5f * v * (1.0f + erff(v * 0.70710678118f)));
          }
        }
      }
    }
}

// ---------------- split qkv -> Qs (scaled), Ks, Vt (transposed, zero-padded) ----------------
__global__ __launch_bounds__(256) void split_qkv(const _Float16* __restrict__ qkv,
                                                 _Float16* __restrict__ Qs,
                                                 _Float16* __restrict__ Ks,
                                                 _Float16* __restrict__ Vt) {
  const size_t idx = (size_t)blockIdx.x * 256 + threadIdx.x;
  const size_t total = (size_t)BH * NPADV * 64;
  if (idx >= total) return;
  const int d = (int)(idx & 63);
  const int n = (int)((idx >> 6) % NPADV);
  const int bh = (int)(idx / ((size_t)NPADV * 64));
  const int b = bh >> 4, h = bh & 15;
  _Float16 v = (_Float16)0.0f;
  if (n < NN) {
    const size_t rowoff = (size_t)(b * NN + n) * (3 * CC);
    const int col = h * DHD + d;
    const _Float16 q = qkv[rowoff + col];
    const _Float16 k = qkv[rowoff + CC + col];
    v = qkv[rowoff + 2 * CC + col];
    Qs[((size_t)bh * NN + n) * 64 + d] = (_Float16)((float)q * SCALE);
    Ks[((size_t)bh * NN + n) * 64 + d] = k;
  }
  Vt[((size_t)bh * 64 + d) * NPADV + n] = v;   // padded tail = 0
}

// ---------------- S = Q*K^T (+mask,+bias) -> attn logits (fp32, d_out) ----------------
__global__ __launch_bounds__(256) void attn_qk(const _Float16* __restrict__ Qs,
                                               const _Float16* __restrict__ Ks,
                                               const float* __restrict__ bias,
                                               const unsigned char* __restrict__ mask,
                                               float* __restrict__ attnO) {
  const int lane = threadIdx.x & 31, wave = threadIdx.x >> 5;
  const int wm = wave >> 2, wn = wave & 3;
  const int bh = blockIdx.z, b = bh >> 4, h = bh & 15;
  const int i0 = blockIdx.y * 64 + wm * 32;
  const int j0 = blockIdx.x * 128 + wn * 32;
  const _Float16* qb = Qs + (size_t)bh * NN * 64;
  const _Float16* kb = Ks + (size_t)bh * NN * 64;
  v8f acc[2][2] = {};
#pragma unroll
  for (int ks = 0; ks < 2; ++ks) {             // K = DH = 64 -> two k-steps
    const v16h a0 = gfrag64(qb, i0, ks * 32, lane);
    const v16h a1 = gfrag64(qb, i0 + 16, ks * 32, lane);
    const v16h b0 = gfrag64(kb, j0, ks * 32, lane);
    const v16h b1 = gfrag64(kb, j0 + 16, ks * 32, lane);
    acc[0][0] = wmma16(a0, b0, acc[0][0]);
    acc[0][1] = wmma16(a0, b1, acc[0][1]);
    acc[1][0] = wmma16(a1, b0, acc[1][0]);
    acc[1][1] = wmma16(a1, b1, acc[1][1]);
  }
#pragma unroll
  for (int ti = 0; ti < 2; ++ti)
#pragma unroll
    for (int tj = 0; tj < 2; ++tj) {
      const int col = j0 + tj * 16 + (lane & 15);
      const int rbase = i0 + ti * 16 + ((lane >> 4) << 3);
#pragma unroll
      for (int r = 0; r < 8; ++r) {
        const int row = rbase + r;
        if (row < NN && col < NN) {
          float s = acc[ti][tj][r];
          if (mask[((size_t)b * NN + row) * NN + col]) s = NEGF;
          s += bias[((size_t)h * NN + row) * NN + col];
          attnO[(((size_t)bh) * NN + row) * NN + col] = s;
        }
      }
    }
}

// ---------------- softmax row stats: max and 1/sum(exp) ----------------
__global__ __launch_bounds__(256) void softmax_stats(const float* __restrict__ attn,
                                                     float* __restrict__ rowmax,
                                                     float* __restrict__ rowinv) {
  __shared__ float red[256];
  const size_t row = blockIdx.x;
  const float* sr = attn + row * (size_t)NN;
  const int t = threadIdx.x;
  float m = -3.4e38f;
  for (int c = t; c < NN; c += 256) m = fmaxf(m, sr[c]);
  red[t] = m;
  __syncthreads();
  for (int st = 128; st > 0; st >>= 1) {
    if (t < st) red[t] = fmaxf(red[t], red[t + st]);
    __syncthreads();
  }
  m = red[0];
  __syncthreads();
  float s = 0.0f;
  for (int c = t; c < NN; c += 256) s += __expf(sr[c] - m);
  red[t] = s;
  __syncthreads();
  for (int st = 128; st > 0; st >>= 1) {
    if (t < st) red[t] += red[t + st];
    __syncthreads();
  }
  if (t == 0) { rowmax[row] = m; rowinv[row] = 1.0f / red[0]; }
}

// ---------------- O = softmax(S) * V  (on-the-fly p, fp16 WMMA) ----------------
__global__ __launch_bounds__(256) void attn_av(const float* __restrict__ attnL,
                                               const float* __restrict__ rowmax,
                                               const float* __restrict__ rowinv,
                                               const _Float16* __restrict__ Vt,
                                               _Float16* __restrict__ o16) {
  __shared__ _Float16 sV[64 * LDS_STRIDE];
  const int tid = threadIdx.x, lane = tid & 31, wave = tid >> 5;
  const int bh = blockIdx.y, b = bh >> 4, h = bh & 15;
  const int r0 = blockIdx.x * 128 + wave * 16;   // 8 waves * 16 rows
  int myrow = r0 + (lane & 15);
  if (myrow > NN - 1) myrow = NN - 1;
  const float rmax = rowmax[(size_t)bh * NN + myrow];
  const float rinv = rowinv[(size_t)bh * NN + myrow];
  const float* srow = attnL + ((size_t)bh * NN + myrow) * NN;
  const _Float16* vb = Vt + (size_t)bh * 64 * NPADV;
  v8f acc[4] = {};
  for (int k0 = 0; k0 < NPADV; k0 += 32) {
    __syncthreads();
    {                                             // stage 64x32 V^T tile
      const int r = tid >> 2, s = tid & 3;
      *(uint4*)(sV + r * LDS_STRIDE + s * 8) =
          *(const uint4*)(vb + (size_t)r * NPADV + k0 + s * 8);
    }
    __syncthreads();
    const int kb = k0 + ((lane >> 4) << 3);
    float p0[8], p1[8];
    if (k0 + 32 <= 1024) {                        // all cols guaranteed < NN
#pragma unroll
      for (int q = 0; q < 8; ++q) {
        p0[q] = __expf(srow[kb + q] - rmax) * rinv;
        p1[q] = __expf(srow[kb + 16 + q] - rmax) * rinv;
      }
    } else {                                      // tail: cols >= NN contribute 0
#pragma unroll
      for (int q = 0; q < 8; ++q) {
        const int c0i = kb + q, c1i = kb + 16 + q;
        p0[q] = (c0i < NN) ? __expf(srow[c0i] - rmax) * rinv : 0.0f;
        p1[q] = (c1i < NN) ? __expf(srow[c1i] - rmax) * rinv : 0.0f;
      }
    }
    union { _Float16 hh[16]; v16h v; } pu;
#pragma unroll
    for (int q = 0; q < 8; ++q) {
      pu.hh[q] = (_Float16)p0[q];
      pu.hh[8 + q] = (_Float16)p1[q];
    }
#pragma unroll
    for (int t4 = 0; t4 < 4; ++t4) {
      const v16h bf = lds_frag(sV + t4 * 16 * LDS_STRIDE, lane);
      acc[t4] = wmma16(pu.v, bf, acc[t4]);
    }
  }
#pragma unroll
  for (int t4 = 0; t4 < 4; ++t4) {
    const int col = h * DHD + t4 * 16 + (lane & 15);
    const int rbase = r0 + ((lane >> 4) << 3);
#pragma unroll
    for (int r = 0; r < 8; ++r) {
      const int row = rbase + r;
      if (row < NN)
        o16[((size_t)b * NN + row) * CC + col] = (_Float16)acc[t4][r];
    }
  }
}

// ---------------- launch ----------------
extern "C" void kernel_launch(void* const* d_in, const int* in_sizes, int n_in,
                              void* d_out, int out_size, void* d_ws, size_t ws_size,
                              hipStream_t stream) {
  (void)in_sizes; (void)n_in; (void)out_size; (void)ws_size;
  const float* x     = (const float*)d_in[0];
  const float* rpb   = (const float*)d_in[1];
  const unsigned char* mask = (const unsigned char*)d_in[2];
  const float* ln1g  = (const float*)d_in[3];
  const float* ln1b  = (const float*)d_in[4];
  const float* qkvw  = (const float*)d_in[5];
  const float* qkvb  = (const float*)d_in[6];
  const float* projw = (const float*)d_in[7];
  const float* projb = (const float*)d_in[8];
  const float* ln2g  = (const float*)d_in[9];
  const float* ln2b  = (const float*)d_in[10];
  const float* fc1w  = (const float*)d_in[11];
  const float* fc1b  = (const float*)d_in[12];
  const float* fc2w  = (const float*)d_in[13];
  const float* fc2b  = (const float*)d_in[14];

  float* xout  = (float*)d_out;                         // [B,N,C]
  float* attnL = xout + (size_t)BB * NN * CC;           // [B,H,N,N] logits

  char* ws = (char*)d_ws;                               // static arena (256B aligned)
  size_t off = 0;
  _Float16* h16    = (_Float16*)(ws + off); off += (size_t)MROWS * CC * 2;
  _Float16* qkv16  = (_Float16*)(ws + off); off += (size_t)MROWS * 3 * CC * 2;
  _Float16* Qs     = (_Float16*)(ws + off); off += (size_t)BH * NN * DHD * 2;
  _Float16* Ks     = (_Float16*)(ws + off); off += (size_t)BH * NN * DHD * 2;
  _Float16* Vt     = (_Float16*)(ws + off); off += (size_t)BH * DHD * NPADV * 2;
  float*    rmax   = (float*)(ws + off);    off += (size_t)SR * 4;
  float*    rinv   = (float*)(ws + off);    off += (size_t)SR * 4;
  _Float16* o16    = (_Float16*)(ws + off); off += (size_t)MROWS * CC * 2;
  float*    x1     = (float*)(ws + off);    off += (size_t)MROWS * CC * 4;
  _Float16* h2     = (_Float16*)(ws + off); off += (size_t)MROWS * CC * 2;
  _Float16* h3     = (_Float16*)(ws + off); off += (size_t)MROWS * HIDD * 2;
  _Float16* wqkvT  = (_Float16*)(ws + off); off += (size_t)3 * CC * CC * 2;
  _Float16* wprojT = (_Float16*)(ws + off); off += (size_t)CC * CC * 2;
  _Float16* wfc1T  = (_Float16*)(ws + off); off += (size_t)CC * HIDD * 2;
  _Float16* wfc2T  = (_Float16*)(ws + off); off += (size_t)CC * HIDD * 2;

  const size_t gemmLds = 4u * TILE_HALVES * 2u;         // 40 KB: A0,B0,A1,B1

  // 1) weight transpose + fp16 convert
  wtrans<<<(3 * CC * CC + 255) / 256, 256, 0, stream>>>(qkvw, wqkvT, CC, 3 * CC);
  wtrans<<<(CC * CC + 255) / 256, 256, 0, stream>>>(projw, wprojT, CC, CC);
  wtrans<<<(CC * HIDD + 255) / 256, 256, 0, stream>>>(fc1w, wfc1T, CC, HIDD);
  wtrans<<<(CC * HIDD + 255) / 256, 256, 0, stream>>>(fc2w, wfc2T, HIDD, CC);

  // 2) LN1 -> h16
  layernorm_k<<<MROWS, 256, 0, stream>>>(x, ln1g, ln1b, h16);

  // 3) qkv = h @ qkv_w + b  (fp16 out)
  gemm16<0><<<dim3(3 * CC / 128, (MROWS + 127) / 128), 256, gemmLds, stream>>>(
      h16, wqkvT, qkvb, nullptr, qkv16, nullptr, MROWS, 3 * CC, CC);

  // 4) split into Qs (scaled), Ks, Vt (transposed + padded)
  split_qkv<<<(int)(((size_t)BH * NPADV * 64 + 255) / 256), 256, 0, stream>>>(
      qkv16, Qs, Ks, Vt);

  // 5) logits = Q*K^T, mask, +rel_pos_bias -> d_out attn section
  attn_qk<<<dim3((NN + 127) / 128, (NN + 63) / 64, BH), 256, 0, stream>>>(
      Qs, Ks, rpb, mask, attnL);

  // 6) softmax row statistics
  softmax_stats<<<SR, 256, 0, stream>>>(attnL, rmax, rinv);

  // 7) O = softmax(S) @ V  -> o16 in [B,N,C] layout
  attn_av<<<dim3((NN + 127) / 128, BH), 256, 0, stream>>>(attnL, rmax, rinv, Vt, o16);

  // 8) x1 = x + o @ proj_w + b   (f32)
  gemm16<1><<<dim3(CC / 128, (MROWS + 127) / 128), 256, gemmLds, stream>>>(
      o16, wprojT, projb, x, nullptr, x1, MROWS, CC, CC);

  // 9) LN2 -> h2
  layernorm_k<<<MROWS, 256, 0, stream>>>(x1, ln2g, ln2b, h2);

  // 10) h3 = gelu(h2 @ fc1_w + b)  (fp16)
  gemm16<2><<<dim3(HIDD / 128, (MROWS + 127) / 128), 256, gemmLds, stream>>>(
      h2, wfc1T, fc1b, nullptr, h3, nullptr, MROWS, HIDD, CC);

  // 11) out = x1 + h3 @ fc2_w + b  -> d_out x section
  gemm16<1><<<dim3(CC / 128, (MROWS + 127) / 128), 256, gemmLds, stream>>>(
      h3, wfc2T, fc2b, x1, nullptr, xout, MROWS, CC, HIDD);
}
